// ScaledDotProductAttention_65120294142184
// MI455X (gfx1250) — compile-verified, matrix operand order checked by reading, not run on hardware
//
#include <hip/hip_runtime.h>

// Problem constants (match reference)
constexpr int Bc = 4, Hc = 16, Nc = 2048, Dc = 64;
#define BQ 64      // query rows per workgroup
#define BK 32      // keys per inner iteration
#define NWAVES 4   // 128 threads, wave32

typedef _Float16 v16h  __attribute__((ext_vector_type(16)));
typedef _Float16 half8 __attribute__((ext_vector_type(8)));
typedef float    v8f   __attribute__((ext_vector_type(8)));

struct FalseT { static constexpr bool value = false; };
struct TrueT  { static constexpr bool value = true;  };

// Load a WMMA A/B operand (16 halves per lane): two contiguous 16B chunks
// at p[0..7] and p[16..23], matching the ISA 16-bit 16x32 per-lane layout.
__device__ __forceinline__ v16h load_ab16(const _Float16* p) {
  half8 lo = *(const half8*)(p);
  half8 hi = *(const half8*)(p + 16);
  return __builtin_shufflevector(lo, hi, 0,1,2,3,4,5,6,7,8,9,10,11,12,13,14,15);
}

// Async global->LDS copy of 16 bytes (CDNA5, tracked by ASYNCcnt).
__device__ __forceinline__ void async_b128(const _Float16* lds_dst,
                                           const _Float16* g_src) {
  unsigned l = (unsigned)(unsigned long long)lds_dst;
  asm volatile("global_load_async_to_lds_b128 %0, %1, off"
               :: "v"(l), "v"(g_src) : "memory");
}

// Raw v_exp_f32 (flush-to-zero for tiny results: exactly what softmax wants).
__device__ __forceinline__ float fast_exp2(float x) {
#if __has_builtin(__builtin_amdgcn_exp2f)
  return __builtin_amdgcn_exp2f(x);
#else
  return exp2f(x);
#endif
}

// xor-permute within each 16-lane half: VALU v_permlane16 if available.
template<unsigned SEL0, unsigned SEL1, int M>
__device__ __forceinline__ float lane_permx(float x) {
#if __has_builtin(__builtin_amdgcn_permlane16)
  int u = __float_as_int(x);
  u = __builtin_amdgcn_permlane16(u, u, (int)SEL0, (int)SEL1, false, false);
  return __int_as_float(u);
#else
  return __shfl_xor(x, M);
#endif
}

__device__ __forceinline__ float red_max16(float t) {
  t = fmaxf(t, lane_permx<0x67452301u, 0xEFCDAB89u, 1>(t));
  t = fmaxf(t, lane_permx<0x54761032u, 0xDCFE98BAu, 2>(t));
  t = fmaxf(t, lane_permx<0x32107654u, 0xBA98FEDCu, 4>(t));
  t = fmaxf(t, lane_permx<0xFEDCBA98u, 0x76543210u, 8>(t));
  return t;
}

// ======================= FAST PATH (needs 48MB ws) =======================

// Prep: Q_hat, K_hat as f16 row-major; V as f16 transposed [bh][d][n].
__global__ void prep_kernel(const float* __restrict__ Q,
                            const float* __restrict__ K,
                            const float* __restrict__ V,
                            _Float16* __restrict__ Qh,
                            _Float16* __restrict__ Kh,
                            _Float16* __restrict__ Vt) {
  int row = blockIdx.x * blockDim.x + threadIdx.x;  // bh*N + n
  if (row >= Bc * Hc * Nc) return;
  int bh = row >> 11;          // /Nc
  int n  = row & (Nc - 1);

  const float* srcs[2] = { Q + (size_t)row * Dc, K + (size_t)row * Dc };
  _Float16*    dsts[2] = { Qh + (size_t)row * Dc, Kh + (size_t)row * Dc };
#pragma unroll
  for (int which = 0; which < 2; ++which) {
    const float4* p4 = (const float4*)srcs[which];
    float s = 0.f;
#pragma unroll
    for (int i = 0; i < Dc / 4; ++i) {
      float4 v = p4[i];
      s += v.x * v.x + v.y * v.y + v.z * v.z + v.w * v.w;
    }
    float inv = 1.0f / (sqrtf(s) + 1e-8f);
    half8* d8 = (half8*)dsts[which];
#pragma unroll
    for (int i = 0; i < Dc / 8; ++i) {
      float4 a = p4[2 * i], b = p4[2 * i + 1];
      half8 h = { (_Float16)(a.x * inv), (_Float16)(a.y * inv),
                  (_Float16)(a.z * inv), (_Float16)(a.w * inv),
                  (_Float16)(b.x * inv), (_Float16)(b.y * inv),
                  (_Float16)(b.z * inv), (_Float16)(b.w * inv) };
      d8[i] = h;
    }
  }
  const float* vrow = V + (size_t)row * Dc;
  _Float16* vt = Vt + (size_t)bh * Dc * Nc + n;
#pragma unroll
  for (int d = 0; d < Dc; ++d) vt[(size_t)d * Nc] = (_Float16)vrow[d];
}

__global__ __launch_bounds__(128) void flash_attn_async(
    const _Float16* __restrict__ Qh, const _Float16* __restrict__ Kh,
    const _Float16* __restrict__ Vt, const float* __restrict__ qk_scale,
    float* __restrict__ out) {
  constexpr int LQ = 72, LK = 72, LV = 48, LP = 40;  // LDS strides (halves)
  __shared__ __align__(16) _Float16 sQ[BQ * LQ];
  __shared__ __align__(16) _Float16 sK[3][BK * LK];     // triple-buffered
  __shared__ __align__(16) _Float16 sVt[3][Dc * LV];    // triple-buffered
  __shared__ __align__(16) _Float16 sP[NWAVES][16 * LP];

  const int i0 = blockIdx.x * BQ;
  const int bh = blockIdx.y;
  const int h  = bh & (Hc - 1);
  const float gl2 = qk_scale[h] * 1.44269504088896340736f;  // exp2 domain

  const _Float16* Qb = Qh + (size_t)bh * Nc * Dc;
  const _Float16* Kb = Kh + (size_t)bh * Nc * Dc;
  const _Float16* Vb = Vt + (size_t)bh * Dc * Nc;

  const int tid  = threadIdx.x;
  const int lane = tid & 31;
  const int wave = tid >> 5;
  const int hid  = lane >> 4;
  const int lr   = lane & 15;
  const int kb   = hid * 8;

  const int njt = i0 / BK + 2;

  // ---- prologue: async-stage Q (8KB) and K/V tile 0 ----
#pragma unroll
  for (int it = 0; it < 4; ++it) {   // 512 x 16B chunks of Q
    int idx = tid + 128 * it;
    int r = idx >> 3, c = idx & 7;
    async_b128(&sQ[r * LQ + c * 8], Qb + (size_t)(i0 + r) * Dc + c * 8);
  }
#pragma unroll
  for (int it = 0; it < 2; ++it) {   // K tile: 256 x 16B
    int idx = tid + 128 * it;
    int r = idx >> 3, c = idx & 7;
    async_b128(&sK[0][r * LK + c * 8], Kb + (size_t)r * Dc + c * 8);
  }
#pragma unroll
  for (int it = 0; it < 2; ++it) {   // V^T tile: 64 d-rows x 4 chunks
    int idx = tid + 128 * it;
    int d = idx >> 2, c = idx & 3;
    async_b128(&sVt[0][d * LV + c * 8], Vb + (size_t)d * Nc + c * 8);
  }

  // All-ones B operand: 5th accumulator tile integrates rowsum(P) via WMMA,
  // broadcast to all 16 lanes of each row half (B columns are identical).
  v16h vones;
#pragma unroll
  for (int i = 0; i < 16; ++i) vones[i] = (_Float16)1.0f;

  float mrow[8];
  v8f o[5];                       // o[0..3]: O tiles; o[4]: softmax denominator
#pragma unroll
  for (int v = 0; v < 8; ++v) mrow[v] = -1e30f;
#pragma unroll
  for (int t = 0; t < 5; ++t)
#pragma unroll
    for (int v = 0; v < 8; ++v) o[t][v] = 0.f;

  v16h aQ0, aQ1;      // hoisted Q operands (loaded once, reused every tile)
  int cur = 0;        // rotating triple-buffer index

  auto tile_step = [&](int jt, auto mc) {
    constexpr bool MASKED = decltype(mc)::value;
    const int nxt = (cur == 2) ? 0 : cur + 1;
    const bool pre = (jt + 1 < njt);
    if (pre) {  // block-uniform: stream next tile into the next buffer
      const int j1 = (jt + 1) * BK;
#pragma unroll
      for (int it = 0; it < 2; ++it) {
        int idx = tid + 128 * it;
        int r = idx >> 3, c = idx & 7;
        async_b128(&sK[nxt][r * LK + c * 8],
                   Kb + (size_t)(j1 + r) * Dc + c * 8);
      }
#pragma unroll
      for (int it = 0; it < 2; ++it) {
        int idx = tid + 128 * it;
        int d = idx >> 2, c = idx & 3;
        async_b128(&sVt[nxt][d * LV + c * 8],
                   Vb + (size_t)d * Nc + j1 + c * 8);
      }
      asm volatile("s_wait_asynccnt 0x4" ::: "memory");  // tile jt is done
    } else {
      asm volatile("s_wait_asynccnt 0x0" ::: "memory");
    }
    // Single barrier/iteration: with 3 buffers, the refill of buf[b] at
    // iteration jt+2 is ordered after this barrier at jt+1, which is after
    // the last read of buf[b] (compute of iteration jt).
    __syncthreads();

    if (jt == 0) {  // Q is resident after the first wait+barrier
      aQ0 = load_ab16(&sQ[(16 * wave + lr) * LQ + kb]);
      aQ1 = load_ab16(&sQ[(16 * wave + lr) * LQ + kb + 32]);
    }

    const int j0 = jt * BK;
    // ---- S = Q_hat * K_hat^T : two 16x16 tiles, K=32 f16 WMMA x2 each ----
    v8f s0, s1;
#pragma unroll
    for (int v = 0; v < 8; ++v) { s0[v] = 0.f; s1[v] = 0.f; }
    {
      v16h b0 = load_ab16(&sK[cur][lr * LK + kb]);
      v16h b1 = load_ab16(&sK[cur][(16 + lr) * LK + kb]);
      s0 = __builtin_amdgcn_wmma_f32_16x16x32_f16(false, aQ0, false, b0,
                                                  (short)0, s0, false, false);
      s1 = __builtin_amdgcn_wmma_f32_16x16x32_f16(false, aQ0, false, b1,
                                                  (short)0, s1, false, false);
      b0 = load_ab16(&sK[cur][lr * LK + kb + 32]);
      b1 = load_ab16(&sK[cur][(16 + lr) * LK + kb + 32]);
      s0 = __builtin_amdgcn_wmma_f32_16x16x32_f16(false, aQ1, false, b0,
                                                  (short)0, s0, false, false);
      s1 = __builtin_amdgcn_wmma_f32_16x16x32_f16(false, aQ1, false, b1,
                                                  (short)0, s1, false, false);
    }

    // ---- scale (exp2 domain), optional causal mask, online softmax ----
    const int colg0 = j0 + lr;
    const int colg1 = j0 + 16 + lr;
    const int rowg0 = i0 + 16 * wave + 8 * hid;
    float x0[8], x1[8];
#pragma unroll
    for (int v = 0; v < 8; ++v) {
      float a0 = s0[v] * gl2, a1 = s1[v] * gl2;
      if (MASKED) {
        int rg = rowg0 + v;
        x0[v] = (colg0 <= rg) ? a0 : -1e30f;
        x1[v] = (colg1 <= rg) ? a1 : -1e30f;
      } else {
        x0[v] = a0; x1[v] = a1;
      }
    }
#pragma unroll
    for (int v = 0; v < 8; ++v) {
      float t = red_max16(fmaxf(x0[v], x1[v]));
      float mnew  = fmaxf(mrow[v], t);
      float alpha = fast_exp2(mrow[v] - mnew);
      mrow[v] = mnew;
      float p0 = fast_exp2(x0[v] - mnew);
      float p1 = fast_exp2(x1[v] - mnew);
      int pr = (v + 8 * hid) * LP + lr;
      sP[wave][pr]      = (_Float16)p0;
      sP[wave][pr + 16] = (_Float16)p1;
#pragma unroll
      for (int t4 = 0; t4 < 5; ++t4) o[t4][v] *= alpha;
    }

    asm volatile("s_wait_dscnt 0x0" ::: "memory");

    // ---- [O | l] += P (16x32) x [V | 1] : 4 LDS B tiles + ones tile ----
    v16h ap = load_ab16(&sP[wave][lr * LP + kb]);
#pragma unroll
    for (int t4 = 0; t4 < 4; ++t4) {
      v16h bv = load_ab16(&sVt[cur][(16 * t4 + lr) * LV + kb]);
      o[t4] = __builtin_amdgcn_wmma_f32_16x16x32_f16(false, ap, false, bv,
                                                     (short)0, o[t4], false, false);
    }
    o[4] = __builtin_amdgcn_wmma_f32_16x16x32_f16(false, ap, false, vones,
                                                  (short)0, o[4], false, false);
    cur = nxt;
  };

  int jt = 0;
  const int njt_full = njt - 2;         // tiles fully inside the causal region
  for (; jt < njt_full; ++jt) tile_step(jt, FalseT{});
  for (; jt < njt;      ++jt) tile_step(jt, TrueT{});

  // ---- epilogue: divide by WMMA-accumulated denominator ----
#pragma unroll
  for (int v = 0; v < 8; ++v) {
    int rg = i0 + 16 * wave + 8 * hid + v;
    float invl = 1.0f / o[4][v];
    float* orow = out + (size_t)bh * Nc * Dc + (size_t)rg * Dc;
#pragma unroll
    for (int t4 = 0; t4 < 4; ++t4) orow[16 * t4 + lr] = o[t4][v] * invl;
  }
}

// ======================= FALLBACK PATH (1MB ws) =======================

__global__ void rownorm_kernel(const float* __restrict__ Q,
                               const float* __restrict__ K,
                               float* __restrict__ invq,
                               float* __restrict__ invk, int rows) {
  int idx = blockIdx.x * blockDim.x + threadIdx.x;
  if (idx >= 2 * rows) return;
  const float* src = (idx < rows) ? (Q + (size_t)idx * Dc)
                                  : (K + (size_t)(idx - rows) * Dc);
  const float4* p4 = (const float4*)src;
  float s = 0.f;
#pragma unroll
  for (int i = 0; i < Dc / 4; ++i) {
    float4 v = p4[i];
    s += v.x * v.x + v.y * v.y + v.z * v.z + v.w * v.w;
  }
  float inv = 1.0f / (sqrtf(s) + 1e-8f);
  if (idx < rows) invq[idx] = inv; else invk[idx - rows] = inv;
}

__global__ __launch_bounds__(128) void flash_attn_fallback(
    const float* __restrict__ Q, const float* __restrict__ K,
    const float* __restrict__ V, const float* __restrict__ qk_scale,
    const float* __restrict__ invq, const float* __restrict__ invk,
    float* __restrict__ out) {
  constexpr int LQ = 72, LK = 72, LV = 40, LP = 40;
  __shared__ __align__(16) _Float16 sQ[BQ * LQ];
  __shared__ __align__(16) _Float16 sK[BK * LK];
  __shared__ __align__(16) _Float16 sVt[Dc * LV];
  __shared__ __align__(16) _Float16 sP[NWAVES][16 * LP];

  const int i0 = blockIdx.x * BQ;
  const int bh = blockIdx.y;
  const int h  = bh & (Hc - 1);
  const size_t base = (size_t)bh * Nc * Dc;
  const float gl2 = qk_scale[h] * 1.44269504088896340736f;

  const int tid  = threadIdx.x;
  const int lane = tid & 31;
  const int wave = tid >> 5;
  const int hid  = lane >> 4;
  const int lr   = lane & 15;
  const int kb   = hid * 8;

  {
    const float* Qb  = Q + base;
    const float* ivq = invq + (size_t)bh * Nc + i0;
#pragma unroll
    for (int it = 0; it < 8; ++it) {
      int idx = tid + 128 * it;
      int r = idx >> 4, c4 = idx & 15;
      float4 v  = ((const float4*)(Qb + (size_t)(i0 + r) * Dc))[c4];
      float inv = ivq[r];
      _Float16* d = &sQ[r * LQ + c4 * 4];
      d[0] = (_Float16)(v.x * inv); d[1] = (_Float16)(v.y * inv);
      d[2] = (_Float16)(v.z * inv); d[3] = (_Float16)(v.w * inv);
    }
  }
  __syncthreads();

  v16h vones;
#pragma unroll
  for (int i = 0; i < 16; ++i) vones[i] = (_Float16)1.0f;

  float mrow[8];
  v8f o[5];
#pragma unroll
  for (int v = 0; v < 8; ++v) mrow[v] = -1e30f;
#pragma unroll
  for (int t = 0; t < 5; ++t)
#pragma unroll
    for (int v = 0; v < 8; ++v) o[t][v] = 0.f;

  const int njt = i0 / BK + 2;
  for (int jt = 0; jt < njt; ++jt) {
    const int j0 = jt * BK;
    {
      const float* Kb  = K + base + (size_t)j0 * Dc;
      const float* Vb  = V + base + (size_t)j0 * Dc;
      const float* ivk = invk + (size_t)bh * Nc + j0;
#pragma unroll
      for (int it = 0; it < 4; ++it) {
        int idx = tid + 128 * it;
        int r = idx >> 4, c4 = idx & 15;
        float4 kv = ((const float4*)(Kb + (size_t)r * Dc))[c4];
        float inv = ivk[r];
        _Float16* dk = &sK[r * LK + c4 * 4];
        dk[0] = (_Float16)(kv.x * inv); dk[1] = (_Float16)(kv.y * inv);
        dk[2] = (_Float16)(kv.z * inv); dk[3] = (_Float16)(kv.w * inv);
        float4 vv = ((const float4*)(Vb + (size_t)r * Dc))[c4];
        int d0 = c4 * 4;
        sVt[(d0 + 0) * LV + r] = (_Float16)vv.x;
        sVt[(d0 + 1) * LV + r] = (_Float16)vv.y;
        sVt[(d0 + 2) * LV + r] = (_Float16)vv.z;
        sVt[(d0 + 3) * LV + r] = (_Float16)vv.w;
      }
    }
    __syncthreads();

    v8f s0, s1;
#pragma unroll
    for (int v = 0; v < 8; ++v) { s0[v] = 0.f; s1[v] = 0.f; }
#pragma unroll
    for (int step = 0; step < 2; ++step) {
      v16h a  = load_ab16(&sQ[(16 * wave + lr) * LQ + kb + 32 * step]);
      v16h b0 = load_ab16(&sK[lr * LK + kb + 32 * step]);
      v16h b1 = load_ab16(&sK[(16 + lr) * LK + kb + 32 * step]);
      s0 = __builtin_amdgcn_wmma_f32_16x16x32_f16(false, a, false, b0,
                                                  (short)0, s0, false, false);
      s1 = __builtin_amdgcn_wmma_f32_16x16x32_f16(false, a, false, b1,
                                                  (short)0, s1, false, false);
    }

    const int colg0 = j0 + lr;
    const int colg1 = j0 + 16 + lr;
    const int rowg0 = i0 + 16 * wave + 8 * hid;
    float x0[8], x1[8];
#pragma unroll
    for (int v = 0; v < 8; ++v) {
      int rg = rowg0 + v;
      float a0 = s0[v] * gl2, a1 = s1[v] * gl2;
      x0[v] = (colg0 <= rg) ? a0 : -1e30f;
      x1[v] = (colg1 <= rg) ? a1 : -1e30f;
    }
#pragma unroll
    for (int v = 0; v < 8; ++v) {
      float t = red_max16(fmaxf(x0[v], x1[v]));
      float mnew  = fmaxf(mrow[v], t);
      float alpha = fast_exp2(mrow[v] - mnew);
      mrow[v] = mnew;
      float p0 = fast_exp2(x0[v] - mnew);
      float p1 = fast_exp2(x1[v] - mnew);
      int pr = (v + 8 * hid) * LP + lr;
      sP[wave][pr]      = (_Float16)p0;
      sP[wave][pr + 16] = (_Float16)p1;
#pragma unroll
      for (int t4 = 0; t4 < 5; ++t4) o[t4][v] *= alpha;
    }

    asm volatile("s_wait_dscnt 0x0" ::: "memory");

    v16h ap = load_ab16(&sP[wave][lr * LP + kb]);
#pragma unroll
    for (int t4 = 0; t4 < 4; ++t4) {
      v16h bv = load_ab16(&sVt[(16 * t4 + lr) * LV + kb]);
      o[t4] = __builtin_amdgcn_wmma_f32_16x16x32_f16(false, ap, false, bv,
                                                     (short)0, o[t4], false, false);
    }
    o[4] = __builtin_amdgcn_wmma_f32_16x16x32_f16(false, ap, false, vones,
                                                  (short)0, o[4], false, false);
    __syncthreads();
  }

#pragma unroll
  for (int v = 0; v < 8; ++v) {
    int rg = i0 + 16 * wave + 8 * hid + v;
    float invl = 1.0f / o[4][v];
    float* orow = out + base + (size_t)rg * Dc;
#pragma unroll
    for (int t4 = 0; t4 < 4; ++t4) orow[16 * t4 + lr] = o[t4][v] * invl;
  }
}

// ======================= host launcher =======================

extern "C" void kernel_launch(void* const* d_in, const int* in_sizes, int n_in,
                              void* d_out, int out_size, void* d_ws, size_t ws_size,
                              hipStream_t stream) {
  (void)in_sizes; (void)n_in; (void)out_size;
  const float* Q  = (const float*)d_in[0];
  const float* K  = (const float*)d_in[1];
  const float* V  = (const float*)d_in[2];
  const float* qk = (const float*)d_in[3];
  // d_in[4] is the tril mask; causality is applied analytically in-kernel.
  float* out = (float*)d_out;

  const size_t rows  = (size_t)Bc * Hc * Nc;       // 131072
  const size_t elems = rows * Dc;                  // 8M
  const size_t need_fast = 3 * elems * sizeof(_Float16);  // 48MB

  dim3 grid(Nc / BQ, Bc * Hc);
  if (ws_size >= need_fast) {
    _Float16* Qh = (_Float16*)d_ws;
    _Float16* Kh = Qh + elems;
    _Float16* Vt = Kh + elems;
    prep_kernel<<<(int)((rows + 255) / 256), 256, 0, stream>>>(Q, K, V, Qh, Kh, Vt);
    flash_attn_async<<<grid, 128, 0, stream>>>(Qh, Kh, Vt, qk, out);
  } else {
    float* invq = (float*)d_ws;
    float* invk = invq + rows;
    int total = (int)(2 * rows);
    rownorm_kernel<<<(total + 255) / 256, 256, 0, stream>>>(Q, K, invq, invk, (int)rows);
    flash_attn_fallback<<<grid, 128, 0, stream>>>(Q, K, V, qk, invq, invk, out);
  }
}
